// ToolCallingModule_54503134986906
// MI455X (gfx1250) — matrix operand order repaired and checked by reading, not compiled, not run on hardware
//
#include <hip/hip_runtime.h>
#include <math.h>

// ---------------------------------------------------------------------------
// Types for WMMA fragments (wave32, gfx1250)
// ---------------------------------------------------------------------------
typedef __attribute__((ext_vector_type(16))) __bf16        bf16x16;
typedef __attribute__((ext_vector_type(8)))  float         f32x8;
typedef __attribute__((ext_vector_type(4)))  float         f32x4;
typedef __attribute__((ext_vector_type(4)))  unsigned int  u32x4;
typedef __attribute__((ext_vector_type(4)))  unsigned short u16x4;

union Frag {
    bf16x16 v;
    u32x4   u[2];
};
struct APair { Frag lo, hi; };   // two 16-row A fragments (rows m0.., m0+16..)

static __device__ __forceinline__ unsigned short f2bf(float f) {
    unsigned int u = __float_as_uint(f);
    unsigned int lsb = (u >> 16) & 1u;
    u += 0x7fffu + lsb;                // round-to-nearest-even
    return (unsigned short)(u >> 16);
}

// gfx1250 async copy: per-lane 16B global -> LDS, tracked by ASYNCcnt.
static __device__ __forceinline__ void async_fill_b128(unsigned lds_off,
                                                       const unsigned short* g) {
    unsigned long long ga = (unsigned long long)(size_t)g;
    asm volatile("global_load_async_to_lds_b128 %0, %1, off"
                 :: "v"(lds_off), "v"(ga) : "memory");
}
static __device__ __forceinline__ void wait_async0() {
    asm volatile("s_wait_asynccnt 0x0" ::: "memory");
}

// ---------------------------------------------------------------------------
// Problem constants
// ---------------------------------------------------------------------------
#define NTOK   16384   // B*S = 4*4096
#define HDIM   2048
#define EDIM   256
#define THDIM  512
#define TDIM   128
#define NREG   64      // NUM_REGISTERED_TOOLS

// ---------------------------------------------------------------------------
// Kernel 1: f32 -> bf16 conversion of hidden states, fused with gate GEMV.
// One wave per token.  should_call = (h . gate_w + gate_b) > 0  (sigmoid>0.5)
// ---------------------------------------------------------------------------
__global__ __launch_bounds__(256)
void k_convert_gate(const float* __restrict__ h,
                    const float* __restrict__ gw,
                    const float* __restrict__ gb,
                    unsigned short* __restrict__ hb,
                    int*   __restrict__ should_i,
                    float* __restrict__ should_f)
{
    const int token = blockIdx.x * (blockDim.x >> 5) + (threadIdx.x >> 5);
    const int lane  = threadIdx.x & 31;
    const float* row = h + (size_t)token * HDIM;
    unsigned short* orow = hb + (size_t)token * HDIM;

    float acc = 0.f;
    #pragma unroll
    for (int i = lane * 4; i < HDIM; i += 128) {
        f32x4 v = *reinterpret_cast<const f32x4*>(row + i);
        f32x4 w = *reinterpret_cast<const f32x4*>(gw + i);
        acc += v.x * w.x + v.y * w.y + v.z * w.z + v.w * w.w;
        u16x4 o;
        o.x = f2bf(v.x); o.y = f2bf(v.y); o.z = f2bf(v.z); o.w = f2bf(v.w);
        *reinterpret_cast<u16x4*>(orow + i) = o;
    }
    #pragma unroll
    for (int off = 16; off; off >>= 1) acc += __shfl_xor(acc, off, 32);
    if (lane == 0) {
        const int s = (acc + gb[0]) > 0.f ? 1 : 0;
        should_i[token] = s;
        should_f[token] = (float)s;
    }
}

// ---------------------------------------------------------------------------
// Kernel 2: weight convert + transpose:  out[n*K + k] = bf16(in[k*N + n])
// ---------------------------------------------------------------------------
__global__ __launch_bounds__(256)
void k_convt(const float* __restrict__ in, unsigned short* __restrict__ out,
             int K, int N)
{
    const int id = blockIdx.x * 256 + threadIdx.x;
    if (id >= K * N) return;
    const int k = id / N;
    const int n = id - k * N;
    out[(size_t)n * K + k] = f2bf(in[id]);
}

// Plain elementwise convert (tool embedding table stays row-major).
__global__ __launch_bounds__(256)
void k_convp(const float* __restrict__ in, unsigned short* __restrict__ out, int n)
{
    const int id = blockIdx.x * 256 + threadIdx.x;
    if (id < n) out[id] = f2bf(in[id]);
}

// ---------------------------------------------------------------------------
// Kernel 3: fused softmax (T=128) + top-3 + valid-slot selection.
// One wave per token; 4 logits per lane.
// ---------------------------------------------------------------------------
__global__ __launch_bounds__(256)
void k_select(const float* __restrict__ logits,
              const int*   __restrict__ should_i,
              float* __restrict__ probs_out,
              int*   __restrict__ sel_tool,
              int*   __restrict__ any_valid)
{
    const int token = blockIdx.x * (blockDim.x >> 5) + (threadIdx.x >> 5);
    const int lane  = threadIdx.x & 31;
    const float* lg = logits + (size_t)token * TDIM;

    f32x4 v = *reinterpret_cast<const f32x4*>(lg + lane * 4);

    // softmax
    float m = fmaxf(fmaxf(v.x, v.y), fmaxf(v.z, v.w));
    #pragma unroll
    for (int off = 16; off; off >>= 1) m = fmaxf(m, __shfl_xor(m, off, 32));
    f32x4 e;
    e.x = __expf(v.x - m); e.y = __expf(v.y - m);
    e.z = __expf(v.z - m); e.w = __expf(v.w - m);
    float s = e.x + e.y + e.z + e.w;
    #pragma unroll
    for (int off = 16; off; off >>= 1) s += __shfl_xor(s, off, 32);
    const float inv = __frcp_rn(s);
    f32x4 p; p.x = e.x * inv; p.y = e.y * inv; p.z = e.z * inv; p.w = e.w * inv;
    *reinterpret_cast<f32x4*>(probs_out + (size_t)token * TDIM + lane * 4) = p;

    // top-3 by logit (same ordering as probs), ties -> smaller index
    float lv[4] = {v.x, v.y, v.z, v.w};
    int picked = 0;
    int top[3];
    #pragma unroll
    for (int k = 0; k < 3; ++k) {
        float bv = -3.402823466e+38f;
        int   bi = TDIM + 1;
        #pragma unroll
        for (int j = 0; j < 4; ++j) {
            if (!(picked & (1 << j)) && lv[j] > bv) { bv = lv[j]; bi = lane * 4 + j; }
        }
        #pragma unroll
        for (int off = 16; off; off >>= 1) {
            float ov = __shfl_xor(bv, off, 32);
            int   oi = __shfl_xor(bi, off, 32);
            if (ov > bv || (ov == bv && oi < bi)) { bv = ov; bi = oi; }
        }
        top[k] = bi;
        if ((bi >> 2) == lane) picked |= 1 << (bi & 3);
    }

    if (lane == 0) {
        const int should = should_i[token];
        int last = -1;
        #pragma unroll
        for (int k = 0; k < 3; ++k)
            if (should && top[k] < NREG) last = k;
        const int av = (last >= 0) ? 1 : 0;
        sel_tool[token]  = top[av ? last : 0];
        any_valid[token] = av;
    }
}

// ---------------------------------------------------------------------------
// Kernel 4: gather selected tool embedding rows (bf16), one wave per token.
// ---------------------------------------------------------------------------
__global__ __launch_bounds__(256)
void k_gather(const unsigned short* __restrict__ tbl,
              const int* __restrict__ sel_tool,
              unsigned short* __restrict__ emb)
{
    const int token = blockIdx.x * (blockDim.x >> 5) + (threadIdx.x >> 5);
    const int lane  = threadIdx.x & 31;
    const unsigned short* src = tbl + (size_t)sel_tool[token] * EDIM;
    unsigned short*       dst = emb + (size_t)token * EDIM;
    *reinterpret_cast<u32x4*>(dst + lane * 8) =
        *reinterpret_cast<const u32x4*>(src + lane * 8);
}

// ---------------------------------------------------------------------------
// Kernel 5: bf16 WMMA GEMM with async-LDS double-buffered B tiles.
//   C[M,N] = act( A[M,K] * Bt[N,K]^T + bias ),  K % 64 == 0.
//   A row-major bf16, optional K-concat (A1: k<K1, A2: k>=K1), both splits
//   multiples of 64.  Bt[n*K + k] pre-transposed bf16 weight.
//   Block = 256 threads = 8 waves; block tile = 256(M) x 64(N); each wave
//   owns 32(M) x 64(N): 8 accumulators, 2 A fragments, 4 shared B fragments
//   -> 8 WMMAs per 32-K step per wave.
//   The 64x32 B tile per K-step is shared by all 8 waves: filled once per
//   block via global_load_async_to_lds_b128 (1 instr/wave = 512B), double
//   buffered in LDS (K manually unrolled by 64 so buffer indices are
//   literals -> ds_load_b128), synced with s_wait_asynccnt + barrier.
//   A fragments are register double-buffered (aE/aO).  grid = (M/256, N/64).
//
// EPI: 0 = bf16 store, 1 = bf16 store + relu,
//      2 = f32 store,  3 = f32 store with where(any_valid, val, hidden)
// ---------------------------------------------------------------------------
template <int EPI>
__global__ __launch_bounds__(256)
void k_gemm(const unsigned short* __restrict__ A1, int lda1, int K1,
            const unsigned short* __restrict__ A2, int lda2, int K2,
            const unsigned short* __restrict__ Bt,
            const float* __restrict__ bias,
            void* __restrict__ Cout, int N,
            const float* __restrict__ hidden,
            const int* __restrict__ anyv, int ldh)
{
    // 2 x 4KB B tiles: [buf][col*4 + k8] as 16B vectors (col<64, k8<4)
    __shared__ u32x4 bsh4[2][256];

    const int K    = K1 + K2;           // multiple of 64
    const int tid  = threadIdx.x;
    const int lane = tid & 31;
    const int wave = tid >> 5;
    const int m0   = blockIdx.x * 256 + wave * 32;
    const int n0   = blockIdx.y * 64;
    const int r    = lane & 15;     // A row / B col / C col within tile
    const int hi   = lane >> 4;     // half-wave id
    const int akoff = hi * 8;       // A K-chunk base (elements)

    // --- B tile cooperative fill: thread t -> col (t>>2), 8-elem chunk (t&3)
    const int fc  = tid >> 2;
    const int fk8 = tid & 3;
    const unsigned short* bfill = Bt + (size_t)(n0 + fc) * K + fk8 * 8;
    const unsigned dst0 = (unsigned)(size_t)&bsh4[0][fc * 4 + fk8];
    const unsigned dst1 = (unsigned)(size_t)&bsh4[1][fc * 4 + fk8];

    // --- per-lane B fragment read base (u32x4 units): col r (+16 per j),
    //     K chunks hi*8..hi*8+7 and 16+hi*8..16+hi*8+7 (matches A).
    const int rb = r * 4 + hi;

    const unsigned short* a1lo = A1 + (size_t)(m0 + r) * lda1;
    const unsigned short* a1hi = A1 + (size_t)(m0 + 16 + r) * lda1;
    const unsigned short* a2lo = A2 ? (A2 + (size_t)(m0 + r) * lda2)
                                    : (const unsigned short*)0;
    const unsigned short* a2hi = A2 ? (A2 + (size_t)(m0 + 16 + r) * lda2)
                                    : (const unsigned short*)0;

    auto loadA = [&](int k) {
        APair a;
        const unsigned short* pl;
        const unsigned short* ph;
        if (k < K1) { pl = a1lo + k;        ph = a1hi + k; }
        else        { pl = a2lo + (k - K1); ph = a2hi + (k - K1); }
        a.lo.u[0] = *reinterpret_cast<const u32x4*>(pl + akoff);
        a.lo.u[1] = *reinterpret_cast<const u32x4*>(pl + akoff + 16);
        a.hi.u[0] = *reinterpret_cast<const u32x4*>(ph + akoff);
        a.hi.u[1] = *reinterpret_cast<const u32x4*>(ph + akoff + 16);
        return a;
    };

    f32x8 acc[8] = {};   // [0..3] rows m0..m0+15, [4..7] rows m0+16..m0+31

    auto compute = [&](int buf, const APair& a) {
        #pragma unroll
        for (int j = 0; j < 4; ++j) {
            Frag b;                        // shared by both M-halves
            b.u[0] = bsh4[buf][rb + j * 64];
            b.u[1] = bsh4[buf][rb + j * 64 + 2];
            acc[j]     = __builtin_amdgcn_wmma_f32_16x16x32_bf16(
                             false, a.lo.v, false, b.v, (short)0, acc[j],
                             false, false);
            acc[4 + j] = __builtin_amdgcn_wmma_f32_16x16x32_bf16(
                             false, a.hi.v, false, b.v, (short)0, acc[4 + j],
                             false, false);
        }
    };

    // prologue: fill buffer 0 with k=0, prefetch A(k=0) into registers
    async_fill_b128(dst0, bfill);
    APair aE = loadA(0);
    APair aO;

    for (int k0 = 0; k0 < K; k0 += 64) {
        // ---- even step: buffer 0, aE ----
        wait_async0();          // my fill of bsh4[0] landed
        __syncthreads();        // everyone's fill done; prior buf0 reads done
        async_fill_b128(dst1, bfill + (k0 + 32));   // overlap next fill
        aO = loadA(k0 + 32);                        // overlap next A
        compute(0, aE);

        // ---- odd step: buffer 1, aO ----
        wait_async0();
        __syncthreads();
        if (k0 + 64 < K) {
            async_fill_b128(dst0, bfill + (k0 + 64));
            aE = loadA(k0 + 64);
        }
        compute(1, aO);
    }

    // Epilogue.  C/D layout: lane col = r, VGPR t holds row hi*8 + t.
    #pragma unroll
    for (int half = 0; half < 2; ++half) {
        #pragma unroll
        for (int j = 0; j < 4; ++j) {
            const int col = n0 + j * 16 + r;
            const float bv = bias ? bias[col] : 0.f;
            #pragma unroll
            for (int t = 0; t < 8; ++t) {
                const int orow = m0 + half * 16 + hi * 8 + t;
                float val = acc[half * 4 + j][t] + bv;
                if (EPI == 1) val = fmaxf(val, 0.f);
                if (EPI <= 1) {
                    ((unsigned short*)Cout)[(size_t)orow * N + col] = f2bf(val);
                } else if (EPI == 2) {
                    ((float*)Cout)[(size_t)orow * N + col] = val;
                } else {
                    const float out = anyv[orow] ? val
                                                 : hidden[(size_t)orow * ldh + col];
                    ((float*)Cout)[(size_t)orow * N + col] = out;
                }
            }
        }
    }
}

// ---------------------------------------------------------------------------
// Host launcher
// ---------------------------------------------------------------------------
extern "C" void kernel_launch(void* const* d_in, const int* in_sizes, int n_in,
                              void* d_out, int out_size, void* d_ws, size_t ws_size,
                              hipStream_t stream)
{
    const float* hs     = (const float*)d_in[0];   // [4,4096,2048]
    const float* tbl    = (const float*)d_in[1];   // [128,256]
    const float* gate_w = (const float*)d_in[2];
    const float* gate_b = (const float*)d_in[3];
    const float* sel_w1 = (const float*)d_in[4];
    const float* sel_b1 = (const float*)d_in[5];
    const float* sel_w2 = (const float*)d_in[6];
    const float* sel_b2 = (const float*)d_in[7];
    const float* pg_w1  = (const float*)d_in[8];
    const float* pg_b1  = (const float*)d_in[9];
    const float* pg_w2  = (const float*)d_in[10];
    const float* pg_b2  = (const float*)d_in[11];
    const float* ri_w1  = (const float*)d_in[12];
    const float* ri_b1  = (const float*)d_in[13];
    const float* ri_w2  = (const float*)d_in[14];
    const float* ri_b2  = (const float*)d_in[15];

    // d_out: enhanced [16384*2048] | tool_probs [16384*128] | should_call [16384]
    float* out_enh    = (float*)d_out;
    float* out_probs  = out_enh + (size_t)NTOK * HDIM;
    float* out_should = out_probs + (size_t)NTOK * TDIM;

    // workspace carve-up
    char* base = (char*)d_ws;
    size_t off = 0;
    auto take = [&](size_t bytes) -> void* {
        off = (off + 255) & ~(size_t)255;
        void* p = base + off;
        off += bytes;
        return p;
    };
    unsigned short* h_bf   = (unsigned short*)take((size_t)NTOK * HDIM * 2);
    unsigned short* w_sel1 = (unsigned short*)take((size_t)THDIM * HDIM * 2);          // [512][2048]
    unsigned short* w_sel2 = (unsigned short*)take((size_t)TDIM * THDIM * 2);          // [128][512]
    unsigned short* w_pg1  = (unsigned short*)take((size_t)THDIM * (HDIM + EDIM) * 2); // [512][2304]
    unsigned short* w_pg2  = (unsigned short*)take((size_t)HDIM * THDIM * 2);          // [2048][512]
    unsigned short* w_ri1  = (unsigned short*)take((size_t)THDIM * (2 * HDIM) * 2);    // [512][4096]
    unsigned short* w_ri2  = (unsigned short*)take((size_t)HDIM * THDIM * 2);          // [2048][512]
    unsigned short* tbl_bf = (unsigned short*)take((size_t)TDIM * EDIM * 2);
    unsigned short* h1     = (unsigned short*)take((size_t)NTOK * THDIM * 2);
    float*          logits = (float*)take((size_t)NTOK * TDIM * 4);
    int*            shld_i = (int*)take((size_t)NTOK * 4);
    int*            sel_t  = (int*)take((size_t)NTOK * 4);
    int*            any_v  = (int*)take((size_t)NTOK * 4);
    unsigned short* emb    = (unsigned short*)take((size_t)NTOK * EDIM * 2);
    unsigned short* p1     = (unsigned short*)take((size_t)NTOK * THDIM * 2);
    unsigned short* tres   = (unsigned short*)take((size_t)NTOK * HDIM * 2);
    unsigned short* r1     = (unsigned short*)take((size_t)NTOK * THDIM * 2);
    (void)ws_size; (void)in_sizes; (void)n_in; (void)out_size;

    const dim3 blk(256);
    const int tok_blocks = NTOK / 8;     // 8 waves (tokens) per block

    // 1. hidden -> bf16, fused gate GEMV
    k_convert_gate<<<tok_blocks, blk, 0, stream>>>(hs, gate_w, gate_b,
                                                   h_bf, shld_i, out_should);

    // 2. weight convert + transpose (Bt[n][k])
    k_convt<<<(HDIM * THDIM + 255) / 256, blk, 0, stream>>>(sel_w1, w_sel1, HDIM, THDIM);
    k_convt<<<(THDIM * TDIM + 255) / 256, blk, 0, stream>>>(sel_w2, w_sel2, THDIM, TDIM);
    k_convt<<<((HDIM + EDIM) * THDIM + 255) / 256, blk, 0, stream>>>(pg_w1, w_pg1, HDIM + EDIM, THDIM);
    k_convt<<<(THDIM * HDIM + 255) / 256, blk, 0, stream>>>(pg_w2, w_pg2, THDIM, HDIM);
    k_convt<<<(2 * HDIM * THDIM + 255) / 256, blk, 0, stream>>>(ri_w1, w_ri1, 2 * HDIM, THDIM);
    k_convt<<<(THDIM * HDIM + 255) / 256, blk, 0, stream>>>(ri_w2, w_ri2, THDIM, HDIM);
    k_convp<<<(TDIM * EDIM + 255) / 256, blk, 0, stream>>>(tbl, tbl_bf, TDIM * EDIM);

    // 3. selector layer 1: h1 = relu(h @ sel_w1 + b1)          [16384 x 512]
    k_gemm<1><<<dim3(NTOK / 256, THDIM / 64), blk, 0, stream>>>(
        h_bf, HDIM, HDIM, nullptr, 0, 0, w_sel1, sel_b1, h1, THDIM,
        nullptr, nullptr, 0);

    // 4. selector layer 2: logits = h1 @ sel_w2 + b2 (f32)     [16384 x 128]
    k_gemm<2><<<dim3(NTOK / 256, TDIM / 64), blk, 0, stream>>>(
        h1, THDIM, THDIM, nullptr, 0, 0, w_sel2, sel_b2, logits, TDIM,
        nullptr, nullptr, 0);

    // 5. softmax + top-3 + valid-slot selection
    k_select<<<tok_blocks, blk, 0, stream>>>(logits, shld_i, out_probs, sel_t, any_v);

    // 6. gather selected tool embeddings
    k_gather<<<tok_blocks, blk, 0, stream>>>(tbl_bf, sel_t, emb);

    // 7. param-gen layer 1: p1 = relu(concat[h, emb] @ pg_w1 + b1)
    k_gemm<1><<<dim3(NTOK / 256, THDIM / 64), blk, 0, stream>>>(
        h_bf, HDIM, HDIM, emb, EDIM, EDIM, w_pg1, pg_b1, p1, THDIM,
        nullptr, nullptr, 0);

    // 8. param-gen layer 2: tool_result = p1 @ pg_w2 + b2 (bf16)
    k_gemm<0><<<dim3(NTOK / 256, HDIM / 64), blk, 0, stream>>>(
        p1, THDIM, THDIM, nullptr, 0, 0, w_pg2, pg_b2, tres, HDIM,
        nullptr, nullptr, 0);

    // 9. integrator layer 1: r1 = relu(concat[h, tool_result] @ ri_w1 + b1)
    k_gemm<1><<<dim3(NTOK / 256, THDIM / 64), blk, 0, stream>>>(
        h_bf, HDIM, HDIM, tres, HDIM, HDIM, w_ri1, ri_b1, r1, THDIM,
        nullptr, nullptr, 0);

    // 10. integrator layer 2 + final select:
    //     enhanced = any_valid ? (r1 @ ri_w2 + b2) : hidden
    k_gemm<3><<<dim3(NTOK / 256, HDIM / 64), blk, 0, stream>>>(
        r1, THDIM, THDIM, nullptr, 0, 0, w_ri2, ri_b2, out_enh, HDIM,
        hs, any_v, HDIM);
}
